// AbsPosAttention_65060164599817
// MI455X (gfx1250) — matrix operand order repaired and checked by reading, not compiled, not run on hardware
//
#include <hip/hip_runtime.h>
#include <hip/hip_bf16.h>

typedef __attribute__((ext_vector_type(16))) __bf16 v16bf;
typedef __attribute__((ext_vector_type(8)))  float  v8f;

#define HEADS 8
#define DK    64
#define DV    192
#define DIMM  1536
#define NTOK  2048
#define BATCH 2
#define QSCALE 0.125f  // 64^-0.5

union FragU { v16bf v; uint4 q[2]; };

__device__ __forceinline__ unsigned short f2bf(float f) {
    unsigned int u = __float_as_uint(f);
    u = (u + 0x7FFFu + ((u >> 16) & 1u)) >> 16;
    return (unsigned short)u;
}

// Async global->LDS 16-byte copy (CDNA5): VDST = LDS byte offset (low 32 bits of
// a generic pointer to LDS), VADDR = 64-bit global address. Tracked by ASYNCcnt.
__device__ __forceinline__ void async_copy_b128(const unsigned short* gsrc,
                                                unsigned short* ldst) {
    const unsigned loff = (unsigned)(size_t)(void*)ldst;
    const unsigned long long ga = (unsigned long long)(size_t)gsrc;
    asm volatile("global_load_async_to_lds_b128 %0, %1, off"
                 :: "v"(loff), "v"(ga) : "memory");
}
__device__ __forceinline__ void async_wait0() {
    asm volatile("s_wait_asynccnt 0x0" ::: "memory");
}

// A-operand fragment (16x32 bf16): lanes 0-15 row m, K {0..7,16..23};
// lanes 16-31 row m, K {8..15,24..31}. base points at tile (m=0,k=0), ld in elems.
__device__ __forceinline__ v16bf load_a_frag(const unsigned short* base, int ld) {
    const int lane = threadIdx.x & 31;
    const int m  = lane & 15;
    const int hi = (lane >> 4) & 1;
    FragU u;
    u.q[0] = *(const uint4*)(base + m * ld + (hi ? 8  : 0));
    u.q[1] = *(const uint4*)(base + m * ld + (hi ? 24 : 16));
    return u.v;
}

// B-operand fragment (32x16 bf16) read from a row-major B^T tile [n][k]:
// lane n holds column n, lanes 0-15 K {0..7,16..23}, lanes 16-31 K {8..15,24..31}.
__device__ __forceinline__ v16bf load_b_frag(const unsigned short* base, int ld) {
    const int lane = threadIdx.x & 31;
    const int n  = lane & 15;
    const int hi = (lane >> 4) & 1;
    FragU u;
    u.q[0] = *(const uint4*)(base + n * ld + (hi ? 8  : 0));
    u.q[1] = *(const uint4*)(base + n * ld + (hi ? 24 : 16));
    return u.v;
}

__device__ __forceinline__ v8f wmma_bf16(v16bf a, v16bf b, v8f c) {
    return __builtin_amdgcn_wmma_f32_16x16x32_bf16(false, a, false, b, (short)0, c,
                                                   false, false);
}

__device__ __forceinline__ v8f vzero8() {
    v8f z;
#pragma unroll
    for (int r = 0; r < 8; ++r) z[r] = 0.0f;
    return z;
}

// ---------------------------------------------------------------- convert ---
__global__ void cvt_bf16_kernel(const float* __restrict__ src,
                                unsigned short* __restrict__ dst, int n) {
    int i = blockIdx.x * blockDim.x + threadIdx.x;
    const int stride = gridDim.x * blockDim.x;
    for (; i < n; i += stride) dst[i] = f2bf(src[i]);
}

// ------------------------------------------------------------------- GEMM ---
// C[M,Nw] = A[M,K] * Bw[K,Nw], bf16 in, fp32 accum.
// MODE 0: store bf16 head-split [B,H,NTOK,d]
// MODE 1: Q epilogue: val*QSCALE + pos[h,tok,dc] + rcb[h,dc], store head-split bf16
// MODE 2: store fp32 [M,Nw] + bias
template <int MODE>
__global__ __launch_bounds__(256) void gemm_kernel(
    const unsigned short* __restrict__ A, const unsigned short* __restrict__ Bw,
    int K, int Nw, int d,
    const float* __restrict__ pos, const float* __restrict__ rcb,
    const float* __restrict__ bias,
    unsigned short* __restrict__ outb, float* __restrict__ outf) {
    __shared__ __align__(16) unsigned short As[128 * 32];
    __shared__ __align__(16) unsigned short Bs[64 * 32];  // B^T tile [n][k]

    const int tid  = threadIdx.x;
    const int w    = tid >> 5;
    const int lane = tid & 31;
    const int rowBase = blockIdx.y * 128;
    const int colBase = blockIdx.x * 64;

    v8f acc[4];
#pragma unroll
    for (int t = 0; t < 4; ++t) acc[t] = vzero8();

    const int nk = K >> 5;
    for (int kc = 0; kc < nk; ++kc) {
        const int k0 = kc << 5;
        // stage A tile 128x32 row-major via async global->LDS (no VGPR roundtrip)
#pragma unroll
        for (int i = 0; i < 2; ++i) {
            const int L = tid + i * 256;
            const int r = L >> 2, ko = (L & 3) << 3;
            async_copy_b128(A + (size_t)(rowBase + r) * K + k0 + ko,
                            As + r * 32 + ko);
        }
        // stage B tile transposed: Bs[n][k] = Bw[k0+k][colBase+n]
        {
            const int k = tid >> 3, n8 = (tid & 7) << 3;
            uint4 t4 = *(const uint4*)(Bw + (size_t)(k0 + k) * Nw + colBase + n8);
            const unsigned short* ts = (const unsigned short*)&t4;
#pragma unroll
            for (int e = 0; e < 8; ++e) Bs[(n8 + e) * 32 + k] = ts[e];
        }
        async_wait0();
        __syncthreads();

        const v16bf a = load_a_frag(As + w * 16 * 32, 32);
#pragma unroll
        for (int t = 0; t < 4; ++t) {
            const v16bf b = load_b_frag(Bs + t * 16 * 32, 32);
            acc[t] = wmma_bf16(a, b, acc[t]);
        }
        __syncthreads();
    }

    // epilogue: C layout -> lanes 0-15: M=r, lanes 16-31: M=r+8, N=lane&15
    const int hi = (lane >> 4) & 1;
    const int nloc = lane & 15;
#pragma unroll
    for (int t = 0; t < 4; ++t) {
        const int ncol = colBase + t * 16 + nloc;
#pragma unroll
        for (int r = 0; r < 8; ++r) {
            const int m = rowBase + w * 16 + r + hi * 8;
            float val = acc[t][r];
            if constexpr (MODE == 2) {
                outf[(size_t)m * Nw + ncol] = val + bias[ncol];
            } else {
                const int b   = m / NTOK, tok = m % NTOK;
                const int h   = ncol / d, dc  = ncol % d;
                if constexpr (MODE == 1)
                    val = val * QSCALE + pos[((size_t)h * NTOK + tok) * DK + dc] +
                          rcb[h * DK + dc];
                outb[(((size_t)(b * HEADS + h) * NTOK) + tok) * d + dc] = f2bf(val);
            }
        }
    }
}

// -------------------------------------------------------- flash attention ---
// Block: 256 threads (8 wave32), handles (b,h) x 128 q-rows; wave w owns 16 rows.
__global__ __launch_bounds__(256) void attn_kernel(
    const unsigned short* __restrict__ Q,   // [B,H,NTOK,64] bf16 (biases fused)
    const unsigned short* __restrict__ Kh,  // [B,H,NTOK,64] bf16
    const unsigned short* __restrict__ Vh,  // [B,H,NTOK,192] bf16
    unsigned short* __restrict__ Ctx) {     // [B,NTOK,H*192] bf16
    __shared__ __align__(16) unsigned short Ks[64 * 64];    // [key][dk]   8 KB
    __shared__ __align__(16) unsigned short Vt[192 * 64];   // [vcol][key] 24 KB
    __shared__ __align__(16) unsigned short Ps[8 * 16 * 64];// per-wave P  16 KB

    const int tid  = threadIdx.x;
    const int w    = tid >> 5;
    const int lane = tid & 31;
    const int bh   = blockIdx.y;
    const int b    = bh / HEADS, h = bh % HEADS;
    const int qBase = blockIdx.x * 128;

    const size_t qoff = ((size_t)bh * NTOK + qBase + w * 16) * DK;
    const v16bf qa0 = load_a_frag(Q + qoff, DK);       // K = 0..31
    const v16bf qa1 = load_a_frag(Q + qoff + 32, DK);  // K = 32..63

    float mi[8], li[8];
    v8f o[12];
#pragma unroll
    for (int r = 0; r < 8; ++r) { mi[r] = -1e30f; li[r] = 0.0f; }
#pragma unroll
    for (int j = 0; j < 12; ++j) o[j] = vzero8();

    for (int kb = 0; kb < NTOK; kb += 64) {
        const unsigned short* kg = Kh + ((size_t)bh * NTOK + kb) * DK;
        const unsigned short* vg = Vh + ((size_t)bh * NTOK + kb) * DV;

        // prefetch next key block toward this WGP while we compute this one
        if (kb + 64 < NTOK) {
            __builtin_prefetch(kg + 64 * DK + tid * 16, 0, 3);
            __builtin_prefetch(vg + 64 * DV + tid * 48, 0, 3);
        }

        // stage K tile row-major [64][64] via async global->LDS
#pragma unroll
        for (int i = 0; i < 2; ++i) {
            const int L = tid + i * 256;
            const int r = L >> 3, ko = (L & 7) << 3;
            async_copy_b128(kg + (size_t)r * DK + ko, Ks + r * 64 + ko);
        }
        // stage V transposed: Vt[n][k] = V[kb+k][n] (overlaps with async K copy)
#pragma unroll
        for (int i = 0; i < 6; ++i) {
            const int L = tid + i * 256;              // 0..1535
            const int r = L / 24, c8 = (L % 24) << 3; // key row, col base
            uint4 t4 = *(const uint4*)(vg + (size_t)r * DV + c8);
            const unsigned short* ts = (const unsigned short*)&t4;
#pragma unroll
            for (int e = 0; e < 8; ++e) Vt[(c8 + e) * 64 + r] = ts[e];
        }
        async_wait0();
        __syncthreads();

        // S = Q * K^T  (16 x 64 logits per wave)
        v8f s[4];
#pragma unroll
        for (int t = 0; t < 4; ++t) {
            v8f z = vzero8();
            z = wmma_bf16(qa0, load_b_frag(Ks + t * 16 * 64, 64), z);
            z = wmma_bf16(qa1, load_b_frag(Ks + t * 16 * 64 + 32, 64), z);
            s[t] = z;
        }

        // online softmax (rows live in 16-lane groups; xor 1/2/4/8 stays in-group)
        float sc[8];
#pragma unroll
        for (int r = 0; r < 8; ++r) {
            float mx = fmaxf(fmaxf(s[0][r], s[1][r]), fmaxf(s[2][r], s[3][r]));
            mx = fmaxf(mx, __shfl_xor(mx, 1, 32));
            mx = fmaxf(mx, __shfl_xor(mx, 2, 32));
            mx = fmaxf(mx, __shfl_xor(mx, 4, 32));
            mx = fmaxf(mx, __shfl_xor(mx, 8, 32));
            const float mn = fmaxf(mi[r], mx);
            sc[r] = __expf(mi[r] - mn);
            mi[r] = mn;
            float rs = 0.0f;
#pragma unroll
            for (int t = 0; t < 4; ++t) {
                const float p = __expf(s[t][r] - mn);
                s[t][r] = p;
                rs += p;
            }
            rs += __shfl_xor(rs, 1, 32);
            rs += __shfl_xor(rs, 2, 32);
            rs += __shfl_xor(rs, 4, 32);
            rs += __shfl_xor(rs, 8, 32);
            li[r] = li[r] * sc[r] + rs;
        }
#pragma unroll
        for (int j = 0; j < 12; ++j)
#pragma unroll
            for (int r = 0; r < 8; ++r) o[j][r] *= sc[r];

        // relayout P: C-format -> row-major LDS tile, then reload as A fragments
        {
            const int hi = (lane >> 4) & 1, nloc = lane & 15;
            unsigned short* pw = Ps + w * 16 * 64;
#pragma unroll
            for (int t = 0; t < 4; ++t)
#pragma unroll
                for (int r = 0; r < 8; ++r)
                    pw[(r + hi * 8) * 64 + t * 16 + nloc] = f2bf(s[t][r]);
            asm volatile("s_wait_dscnt 0" ::: "memory");
            const v16bf p0 = load_a_frag(pw, 64);
            const v16bf p1 = load_a_frag(pw + 32, 64);
#pragma unroll
            for (int j = 0; j < 12; ++j) {
                o[j] = wmma_bf16(p0, load_b_frag(Vt + j * 16 * 64, 64), o[j]);
                o[j] = wmma_bf16(p1, load_b_frag(Vt + j * 16 * 64 + 32, 64), o[j]);
            }
        }
        __syncthreads();
    }

    // normalize and store context [B, NTOK, H*DV]
    const int hi = (lane >> 4) & 1, nloc = lane & 15;
    float rl[8];
#pragma unroll
    for (int r = 0; r < 8; ++r) rl[r] = 1.0f / li[r];
#pragma unroll
    for (int j = 0; j < 12; ++j)
#pragma unroll
        for (int r = 0; r < 8; ++r) {
            const int tok = qBase + w * 16 + r + hi * 8;
            Ctx[((size_t)(b * NTOK + tok)) * (HEADS * DV) + h * DV + j * 16 + nloc] =
                f2bf(o[j][r] * rl[r]);
        }
}

// ----------------------------------------------------------------- launch ---
extern "C" void kernel_launch(void* const* d_in, const int* in_sizes, int n_in,
                              void* d_out, int out_size, void* d_ws, size_t ws_size,
                              hipStream_t stream) {
    const float* x   = (const float*)d_in[0];
    const float* Wq  = (const float*)d_in[1];
    const float* Wk  = (const float*)d_in[2];
    const float* Wv  = (const float*)d_in[3];
    const float* Wo  = (const float*)d_in[4];
    const float* bo  = (const float*)d_in[5];
    const float* pos = (const float*)d_in[6];
    const float* rcb = (const float*)d_in[7];
    float* out = (float*)d_out;

    char* ws = (char*)d_ws;
    size_t off = 0;
    auto alloc = [&](size_t bytes) -> unsigned short* {
        unsigned short* p = (unsigned short*)(ws + off);
        off = (off + bytes + 255) & ~(size_t)255;
        return p;
    };
    const int M = BATCH * NTOK;                        // 4096 rows
    unsigned short* xb  = alloc((size_t)M * DIMM * 2);
    unsigned short* Wqb = alloc((size_t)DIMM * (HEADS * DK) * 2);
    unsigned short* Wkb = alloc((size_t)DIMM * (HEADS * DK) * 2);
    unsigned short* Wvb = alloc((size_t)DIMM * (HEADS * DV) * 2);
    unsigned short* Wob = alloc((size_t)(HEADS * DV) * DIMM * 2);
    unsigned short* Qb  = alloc((size_t)M * (HEADS * DK) * 2);
    unsigned short* Kb  = alloc((size_t)M * (HEADS * DK) * 2);
    unsigned short* Vb  = alloc((size_t)M * (HEADS * DV) * 2);
    unsigned short* Ctx = alloc((size_t)M * (HEADS * DV) * 2);

    cvt_bf16_kernel<<<512, 256, 0, stream>>>(x, xb, M * DIMM);
    cvt_bf16_kernel<<<256, 256, 0, stream>>>(Wq, Wqb, DIMM * HEADS * DK);
    cvt_bf16_kernel<<<256, 256, 0, stream>>>(Wk, Wkb, DIMM * HEADS * DK);
    cvt_bf16_kernel<<<512, 256, 0, stream>>>(Wv, Wvb, DIMM * HEADS * DV);
    cvt_bf16_kernel<<<512, 256, 0, stream>>>(Wo, Wob, HEADS * DV * DIMM);

    dim3 gq((HEADS * DK) / 64, M / 128);   // (8, 32)
    gemm_kernel<1><<<gq, 256, 0, stream>>>(xb, Wqb, DIMM, HEADS * DK, DK,
                                           pos, rcb, nullptr, Qb, nullptr);
    gemm_kernel<0><<<gq, 256, 0, stream>>>(xb, Wkb, DIMM, HEADS * DK, DK,
                                           nullptr, nullptr, nullptr, Kb, nullptr);
    dim3 gv((HEADS * DV) / 64, M / 128);   // (24, 32)
    gemm_kernel<0><<<gv, 256, 0, stream>>>(xb, Wvb, DIMM, HEADS * DV, DV,
                                           nullptr, nullptr, nullptr, Vb, nullptr);

    dim3 ga(NTOK / 128, BATCH * HEADS);    // (16, 16)
    attn_kernel<<<ga, 256, 0, stream>>>(Qb, Kb, Vb, Ctx);

    dim3 go(DIMM / 64, M / 128);           // (24, 32)
    gemm_kernel<2><<<go, 256, 0, stream>>>(Ctx, Wob, HEADS * DV, DIMM, 0,
                                           nullptr, nullptr, bo, nullptr, out);
}
